// RBFKANScorer_6837587935654
// MI455X (gfx1250) — matrix operand order repaired and checked by reading, not compiled, not run on hardware
//
#include <hip/hip_runtime.h>
#include <hip/hip_bf16.h>
#include <cstdint>

typedef __attribute__((ext_vector_type(16))) __bf16       v16bf;
typedef __attribute__((ext_vector_type(8)))  float        v8f;
typedef __attribute__((ext_vector_type(8)))  unsigned int v8u;

#define DIM_C 1024
#define HID_C 64
#define NB_C  64
#define ROWS_PER_WAVE 16
#define WAVES_PER_WG  8
#define ROWS_PER_WG   (ROWS_PER_WAVE * WAVES_PER_WG)

// ---- LDS layout (word offsets). Row stride 516 = 512 data words + 4 pad
// so that lane n -> bank (4n + k) % 64 is conflict-free across 16 lanes.
#define WROW_STRIDE 516
#define WHI_WORDS  (HID_C * WROW_STRIDE)
#define PAR_STRIDE 65          // 64 + 1 pad -> bank (h + k) % 64, conflict-free
#define OFF_WHI 0
#define OFF_WLO (OFF_WHI + WHI_WORDS)
#define OFF_C   (OFF_WLO + WHI_WORDS)
#define OFF_RW  (OFF_C  + HID_C * PAR_STRIDE)
#define OFF_OW  (OFF_RW + HID_C * PAR_STRIDE)
#define LDS_WORDS (OFF_OW + HID_C * PAR_STRIDE)
#define LDS_BYTES (LDS_WORDS * 4)   // 314,112 B of the 320KB WGP LDS

__device__ __forceinline__ unsigned int bf16_rne_bits(float a) {
  unsigned int u = __float_as_uint(a);
  return (u + (0x7FFFu + ((u >> 16) & 1u))) >> 16;
}
// pack two floats as adjacent bf16 (hi split), return residual pair in lopack
__device__ __forceinline__ unsigned int pack_split(float f0, float f1,
                                                   unsigned int& lopack) {
  unsigned int h0 = bf16_rne_bits(f0), h1 = bf16_rne_bits(f1);
  float l0 = f0 - __uint_as_float(h0 << 16);
  float l1 = f1 - __uint_as_float(h1 << 16);
  lopack = bf16_rne_bits(l0) | (bf16_rne_bits(l1) << 16);
  return h0 | (h1 << 16);
}

__global__ void rbf_kan_fused(const float* __restrict__ x,
                              const float* __restrict__ in_w,
                              const float* __restrict__ in_b,
                              const float* __restrict__ centers,
                              const float* __restrict__ log_widths,
                              const float* __restrict__ out_weight,
                              const float* __restrict__ out_bias,
                              float* __restrict__ out) {
  extern __shared__ __attribute__((aligned(16))) unsigned int lds[];
  unsigned int* whi = lds + OFF_WHI;
  unsigned int* wlo = lds + OFF_WLO;
  float* ldc = (float*)(lds + OFF_C);
  float* ldr = (float*)(lds + OFF_RW);
  float* ldw = (float*)(lds + OFF_OW);

  const int tid = threadIdx.x;

  // ---------- stage: in_w -> bf16 hi/lo packed fragments; RBF params ----------
  for (int p = tid; p < HID_C * DIM_C / 2; p += blockDim.x) {
    const float2 w2 = ((const float2*)in_w)[p];   // contiguous pairs, row-major
    const int n = p >> 9;                         // 512 pairs per hidden row
    const int j = p & 511;
    unsigned int lop;
    unsigned int hip_ = pack_split(w2.x, w2.y, lop);
    whi[n * WROW_STRIDE + j] = hip_;
    wlo[n * WROW_STRIDE + j] = lop;
  }
  for (int idx = tid; idx < HID_C * NB_C; idx += blockDim.x) {
    const int h = idx >> 6, k = idx & 63;
    const float sp = __logf(1.0f + __expf(log_widths[idx])) + 0.001f;
    ldr[h * PAR_STRIDE + k] = 1.0f / sp;
    ldc[h * PAR_STRIDE + k] = centers[idx];
    ldw[h * PAR_STRIDE + k] = out_weight[idx];
  }
  __syncthreads();

  // ---------- GEMM: z[16 rows x 64 hidden] via split-bf16 WMMA ----------
  const int wave = tid >> 5;
  const int lane = tid & 31;
  const int nl   = lane & 15;   // row (A) / col (B,C) within tile
  const int c0   = lane >> 4;   // lane half
  const int m0   = blockIdx.x * ROWS_PER_WG + wave * ROWS_PER_WAVE;
  const float* xrow = x + (size_t)(m0 + nl) * DIM_C;
  const int koffA = 8 * c0;     // A fragment: K in [8c0,8c0+8) U [16+8c0,+8)

  v8f acc[4] = {};              // 4 hidden tiles of 16

  float4 a0, a1, a2, a3;
  {
    const float4* p0 = (const float4*)(xrow + koffA);
    a0 = p0[0]; a1 = p0[1];
    const float4* p1 = (const float4*)(xrow + 16 + koffA);
    a2 = p1[0]; a3 = p1[1];
  }

  for (int kk = 0; kk < DIM_C / 32; ++kk) {
    // double-buffer next A tile
    float4 n0 = a0, n1 = a1, n2 = a2, n3 = a3;
    if (kk + 1 < DIM_C / 32) {
      const int k0 = (kk + 1) * 32;
      const float4* p0 = (const float4*)(xrow + k0 + koffA);
      n0 = p0[0]; n1 = p0[1];
      const float4* p1 = (const float4*)(xrow + k0 + 16 + koffA);
      n2 = p1[0]; n3 = p1[1];
    }

    // split-convert current A tile (ISA 16-bit A 16x32 lane layout)
    const float f[16] = {a0.x, a0.y, a0.z, a0.w, a1.x, a1.y, a1.z, a1.w,
                         a2.x, a2.y, a2.z, a2.w, a3.x, a3.y, a3.z, a3.w};
    v8u Ah, Al;
#pragma unroll
    for (int v = 0; v < 8; ++v) {
      unsigned int lop;
      Ah[v] = pack_split(f[2 * v], f[2 * v + 1], lop);
      Al[v] = lop;
    }
    const v16bf ahi = __builtin_bit_cast(v16bf, Ah);
    const v16bf alo = __builtin_bit_cast(v16bf, Al);

    // B fragment: lane half c0 covers K = k0+8c0 .. +7 and k0+16+8c0 .. +7
    const int wword = kk * 16 + 4 * c0;
#pragma unroll
    for (int t = 0; t < 4; ++t) {
      const int n = t * 16 + nl;
      const uint4* bh = (const uint4*)(whi + n * WROW_STRIDE + wword);
      const uint4* bl = (const uint4*)(wlo + n * WROW_STRIDE + wword);
      const uint4 h0 = bh[0], h1 = bh[2];
      const uint4 l0 = bl[0], l1 = bl[2];
      v8u Bh = {h0.x, h0.y, h0.z, h0.w, h1.x, h1.y, h1.z, h1.w};
      v8u Bl = {l0.x, l0.y, l0.z, l0.w, l1.x, l1.y, l1.z, l1.w};
      const v16bf bhi = __builtin_bit_cast(v16bf, Bh);
      const v16bf blo = __builtin_bit_cast(v16bf, Bl);
      v8f c = acc[t];
      c = __builtin_amdgcn_wmma_f32_16x16x32_bf16(false, ahi, false, bhi,
                                                  (short)0, c, false, false);
      c = __builtin_amdgcn_wmma_f32_16x16x32_bf16(false, alo, false, bhi,
                                                  (short)0, c, false, false);
      c = __builtin_amdgcn_wmma_f32_16x16x32_bf16(false, ahi, false, blo,
                                                  (short)0, c, false, false);
      acc[t] = c;
    }
    a0 = n0; a1 = n1; a2 = n2; a3 = n3;
  }

  // ---------- RBF expansion + weighted reduction (z stays in C-layout) ----------
  // C layout: VGPR i -> row M = i + 8*c0 ; lane -> hidden h = t*16 + nl
  float s[8] = {0, 0, 0, 0, 0, 0, 0, 0};
#pragma unroll
  for (int t = 0; t < 4; ++t) {
    const int h = t * 16 + nl;
    const float zb = in_b[h];
    float zr[8];
#pragma unroll
    for (int i = 0; i < 8; ++i) zr[i] = acc[t][i] + zb;
    const float* pc = ldc + h * PAR_STRIDE;
    const float* pr = ldr + h * PAR_STRIDE;
    const float* pw = ldw + h * PAR_STRIDE;
#pragma unroll 4
    for (int k = 0; k < NB_C; ++k) {
      const float cc = pc[k], rr = pr[k], ww = pw[k];
#pragma unroll
      for (int i = 0; i < 8; ++i) {
        const float u = (zr[i] - cc) * rr;
        s[i] = fmaf(__expf(-(u * u)), ww, s[i]);
      }
    }
  }

  // reduce the 16 h-residues across each 16-lane half (stays within half)
#pragma unroll
  for (int off = 1; off < 16; off <<= 1) {
#pragma unroll
    for (int i = 0; i < 8; ++i) s[i] += __shfl_xor(s[i], off, 32);
  }

  if (nl == 0) {
    const float ob = out_bias[0];
    const int mbase = m0 + 8 * c0;
#pragma unroll
    for (int i = 0; i < 8; ++i) out[mbase + i] = s[i] + ob;
  }
}

extern "C" void kernel_launch(void* const* d_in, const int* in_sizes, int n_in,
                              void* d_out, int out_size, void* d_ws, size_t ws_size,
                              hipStream_t stream) {
  const float* x          = (const float*)d_in[0];
  const float* in_w       = (const float*)d_in[1];
  const float* in_b       = (const float*)d_in[2];
  const float* centers    = (const float*)d_in[3];
  const float* log_widths = (const float*)d_in[4];
  const float* out_weight = (const float*)d_in[5];
  const float* out_bias   = (const float*)d_in[6];
  float* out = (float*)d_out;

  const int rows = in_sizes[0] / DIM_C;      // B*T = 65536
  const int grid = rows / ROWS_PER_WG;       // 512 workgroups

  (void)hipFuncSetAttribute((const void*)rbf_kan_fused,
                            hipFuncAttributeMaxDynamicSharedMemorySize,
                            LDS_BYTES);
  rbf_kan_fused<<<grid, WAVES_PER_WG * 32, LDS_BYTES, stream>>>(
      x, in_w, in_b, centers, log_widths, out_weight, out_bias, out);
}
